// CrossSpectralAttention_65575560675805
// MI455X (gfx1250) — matrix operand order repaired and checked by reading, not compiled, not run on hardware
//
#include <hip/hip_runtime.h>
#include <hip/hip_bf16.h>

// ---------------------------------------------------------------------------
// CrossSpectralAttention on MI455X (gfx1250), wave32, v_wmma_f32_16x16x32_f16
// ---------------------------------------------------------------------------
#define BATCH 2
#define CCH   256
#define HH    48
#define WW    48
#define NTOK  (HH * WW)          // 2304
#define NHEAD 8
#define HDIM  32
#define SCALE 0.17677669529663687f   // 32^-0.5

typedef _Float16 v16h __attribute__((ext_vector_type(16)));
typedef _Float16 v8h  __attribute__((ext_vector_type(8)));
typedef float    v8f  __attribute__((ext_vector_type(8)));

union V16 { v16h v; v8h h[2]; };

__device__ __forceinline__ v8f wmma_f16(v16h a, v16h b, v8f c) {
  // (neg_a, A, neg_b, B, c_mod, C, reuse_a, reuse_b)
  return __builtin_amdgcn_wmma_f32_16x16x32_f16(false, a, false, b, (short)0, c,
                                                false, false);
}

// --------------------------- kernel 0a: weights f32 -> f16 ------------------
__global__ void csa_cvt_weights(const float* __restrict__ Wq,
                                const float* __restrict__ Wk,
                                const float* __restrict__ Wv,
                                const float* __restrict__ Wo,
                                _Float16* __restrict__ W16) {
  int idx = blockIdx.x * blockDim.x + threadIdx.x;   // 4 * 65536 threads
  int which = idx >> 16;
  int off   = idx & 0xFFFF;
  const float* src = (which == 0) ? Wq : (which == 1) ? Wk : (which == 2) ? Wv : Wo;
  W16[idx] = (_Float16)src[off];
}

// --------------------------- kernel 0b: x -> x^T f16 [b][n][c] --------------
__global__ void csa_transpose_x(const float* __restrict__ x,
                                _Float16* __restrict__ Xt) {
  int idx = blockIdx.x * blockDim.x + threadIdx.x;   // BATCH*NTOK*CCH threads
  int b = idx / (NTOK * CCH);
  int r = idx - b * (NTOK * CCH);
  int n = r / CCH;
  int c = r - n * CCH;
  Xt[idx] = (_Float16)x[((size_t)b * CCH + c) * NTOK + n];
}

// --------------------------- kernel 1: fused QKV projection -----------------
// Each wave computes a 16(n) x 32(o) tile: one shared A fetch, two independent
// accumulator chains. Q[b][n][c] pre-scaled; K[b][n][c]; V transposed [b][c][n].
__global__ void csa_qkv_proj(const _Float16* __restrict__ Xt,
                             const _Float16* __restrict__ W16,  // Wq|Wk|Wv
                             const float* __restrict__ bq,
                             const float* __restrict__ bk,
                             const float* __restrict__ bv,
                             _Float16* __restrict__ Qb,
                             _Float16* __restrict__ Kb,
                             _Float16* __restrict__ Vb) {
  const int NT = NTOK / 16;                            // 144
  int wid  = blockIdx.x * (blockDim.x >> 5) + (threadIdx.x >> 5);
  int lane = threadIdx.x & 31;
  int l15  = lane & 15;
  int hb   = lane >> 4;

  int nt = wid % NT;  wid /= NT;
  int ot = wid % (CCH / 32); wid /= (CCH / 32);        // 8 o-pairs
  int proj = wid % 3;
  int b    = wid / 3;

  int nbase = nt * 16, obase = ot * 32;
  const _Float16* arow  = Xt + ((size_t)b * NTOK + nbase + l15) * CCH;
  const _Float16* wbase = W16 + (size_t)proj * CCH * CCH;
  const _Float16* brow0 = wbase + (size_t)(obase + l15) * CCH;
  const _Float16* brow1 = wbase + (size_t)(obase + 16 + l15) * CCH;

  v8f acc0 = {}, acc1 = {};
#pragma unroll
  for (int kk = 0; kk < CCH; kk += 32) {
    V16 a, bm0, bm1;
    a.h[0]   = *(const v8h*)(arow + kk + hb * 16);      // K = 0..7  (+16 hi half)
    a.h[1]   = *(const v8h*)(arow + kk + hb * 16 + 8);  // K = 8..15 (+16)
    bm0.h[0] = *(const v8h*)(brow0 + kk + hb * 16);     // K = 0..15 | 16..31
    bm0.h[1] = *(const v8h*)(brow0 + kk + hb * 16 + 8);
    bm1.h[0] = *(const v8h*)(brow1 + kk + hb * 16);
    bm1.h[1] = *(const v8h*)(brow1 + kk + hb * 16 + 8);
    acc0 = wmma_f16(a.v, bm0.v, acc0);
    acc1 = wmma_f16(a.v, bm1.v, acc1);
  }

  int o0 = obase + l15, o1 = obase + 16 + l15;
  const float* biasp = (proj == 0) ? bq : (proj == 1) ? bk : bv;
  float bias0 = biasp[o0], bias1 = biasp[o1];

  if (proj == 0) {
#pragma unroll
    for (int r = 0; r < 8; ++r) {
      int n = nbase + r + 8 * hb;
      size_t base = ((size_t)b * NTOK + n) * CCH;
      Qb[base + o0] = (_Float16)((acc0[r] + bias0) * SCALE);
      Qb[base + o1] = (_Float16)((acc1[r] + bias1) * SCALE);
    }
  } else if (proj == 1) {
#pragma unroll
    for (int r = 0; r < 8; ++r) {
      int n = nbase + r + 8 * hb;
      size_t base = ((size_t)b * NTOK + n) * CCH;
      Kb[base + o0] = (_Float16)(acc0[r] + bias0);
      Kb[base + o1] = (_Float16)(acc1[r] + bias1);
    }
  } else {
#pragma unroll
    for (int r = 0; r < 8; ++r) {
      int n = nbase + r + 8 * hb;
      Vb[((size_t)b * CCH + o0) * NTOK + n] = (_Float16)(acc0[r] + bias0);
      Vb[((size_t)b * CCH + o1) * NTOK + n] = (_Float16)(acc1[r] + bias1);
    }
  }
}

// --------------------------- kernel 2: flash attention ----------------------
// 4 waves/block, 16 query rows per wave, stream 64 keys/iter.
__global__ void csa_attention(const _Float16* __restrict__ Qb,
                              const _Float16* __restrict__ Kb,
                              const _Float16* __restrict__ Vb,
                              _Float16* __restrict__ Ob) {
  __shared__ __align__(32) _Float16 Pl[4][16][64];   // per-wave P tile, 8 KB

  int w    = threadIdx.x >> 5;
  int lane = threadIdx.x & 31;
  int l15  = lane & 15;
  int hb   = lane >> 4;

  int bh = blockIdx.y;
  int b  = bh >> 3;
  int h  = bh & 7;
  int nbase = blockIdx.x * 64 + w * 16;

  // Q A-tile: rows nbase..nbase+15, K-dim = head_dim = 32, loaded once
  const _Float16* qrow = Qb + ((size_t)b * NTOK + nbase + l15) * CCH + h * HDIM;
  V16 aq;
  aq.h[0] = *(const v8h*)(qrow + hb * 16);
  aq.h[1] = *(const v8h*)(qrow + hb * 16 + 8);

  v8f acc0 = {}, acc1 = {};         // output dims 0..15 and 16..31
  float rmax[8], rsum[8];
#pragma unroll
  for (int r = 0; r < 8; ++r) { rmax[r] = -1e30f; rsum[r] = 0.0f; }

  for (int kt = 0; kt < NTOK; kt += 64) {
    // ---- S = Q . K^T for 64 keys (4 independent WMMAs) ----
    v8f s[4];
#pragma unroll
    for (int j = 0; j < 4; ++j) {
      const _Float16* krow =
          Kb + ((size_t)b * NTOK + kt + j * 16 + l15) * CCH + h * HDIM;
      V16 bk;
      bk.h[0] = *(const v8h*)(krow + hb * 16);
      bk.h[1] = *(const v8h*)(krow + hb * 16 + 8);
      v8f z = {};
      s[j] = wmma_f16(aq.v, bk.v, z);
    }

    // ---- online softmax update; stage P (f16) into LDS for layout change ---
#pragma unroll
    for (int r = 0; r < 8; ++r) {
      float m = fmaxf(fmaxf(s[0][r], s[1][r]), fmaxf(s[2][r], s[3][r]));
      m = fmaxf(m, __shfl_xor(m, 1));
      m = fmaxf(m, __shfl_xor(m, 2));
      m = fmaxf(m, __shfl_xor(m, 4));
      m = fmaxf(m, __shfl_xor(m, 8));
      float nm = fmaxf(rmax[r], m);
      float sc = __expf(rmax[r] - nm);
      rmax[r] = nm;
      rsum[r] *= sc;
      acc0[r] *= sc;
      acc1[r] *= sc;

      float psum = 0.0f;
#pragma unroll
      for (int j = 0; j < 4; ++j) {
        float p = __expf(s[j][r] - nm);
        psum += p;
        Pl[w][r + 8 * hb][j * 16 + l15] = (_Float16)p;   // row-major 16x64
      }
      psum += __shfl_xor(psum, 1);
      psum += __shfl_xor(psum, 2);
      psum += __shfl_xor(psum, 4);
      psum += __shfl_xor(psum, 8);
      rsum[r] += psum;
    }
    __syncthreads();

    // ---- O += P . V (two 32-key chunks x two 16-dim tiles = 4 WMMAs) ----
#pragma unroll
    for (int hk = 0; hk < 2; ++hk) {
      V16 pa;
      pa.h[0] = *(const v8h*)(&Pl[w][l15][hk * 32 + hb * 16]);
      pa.h[1] = *(const v8h*)(&Pl[w][l15][hk * 32 + hb * 16 + 8]);

      const _Float16* vrow0 =
          Vb + ((size_t)b * CCH + h * HDIM + l15) * NTOK + kt + hk * 32 + hb * 16;
      V16 bv0;
      bv0.h[0] = *(const v8h*)(vrow0);
      bv0.h[1] = *(const v8h*)(vrow0 + 8);
      acc0 = wmma_f16(pa.v, bv0.v, acc0);

      const _Float16* vrow1 = vrow0 + (size_t)16 * NTOK;   // dims 16..31
      V16 bv1;
      bv1.h[0] = *(const v8h*)(vrow1);
      bv1.h[1] = *(const v8h*)(vrow1 + 8);
      acc1 = wmma_f16(pa.v, bv1.v, acc1);
    }
    __syncthreads();
  }

  // ---- normalize and store O[b][n][c] (f16, row-major over c) ----
#pragma unroll
  for (int r = 0; r < 8; ++r) {
    float inv = 1.0f / rsum[r];
    int n = nbase + r + 8 * hb;
    size_t base = ((size_t)b * NTOK + n) * CCH + h * HDIM;
    Ob[base + l15]      = (_Float16)(acc0[r] * inv);
    Ob[base + 16 + l15] = (_Float16)(acc1[r] * inv);
  }
}

// --------------------------- kernel 3: output projection --------------------
// Each wave computes a 32(o) x 16(n) tile: shared B fetch (attention output),
// two A tiles (Wo rows), two independent accumulator chains. f32 output.
__global__ void csa_out_proj(const _Float16* __restrict__ Ob,
                             const _Float16* __restrict__ Wo16,
                             const float* __restrict__ bo,
                             float* __restrict__ out) {
  const int NT = NTOK / 16;
  int wid  = blockIdx.x * (blockDim.x >> 5) + (threadIdx.x >> 5);
  int lane = threadIdx.x & 31;
  int l15  = lane & 15;
  int hb   = lane >> 4;

  int nt = wid % NT;  wid /= NT;
  int ot = wid % (CCH / 32);
  int b  = wid / (CCH / 32);

  int nbase = nt * 16, obase = ot * 32;
  const _Float16* arow0 = Wo16 + (size_t)(obase + l15) * CCH;          // A = Wo
  const _Float16* arow1 = Wo16 + (size_t)(obase + 16 + l15) * CCH;
  const _Float16* brow  = Ob + ((size_t)b * NTOK + nbase + l15) * CCH; // B = O^T

  v8f acc0 = {}, acc1 = {};
#pragma unroll
  for (int kk = 0; kk < CCH; kk += 32) {
    V16 a0, a1, bm;
    a0.h[0] = *(const v8h*)(arow0 + kk + hb * 8);
    a0.h[1] = *(const v8h*)(arow0 + kk + 16 + hb * 8);
    a1.h[0] = *(const v8h*)(arow1 + kk + hb * 8);
    a1.h[1] = *(const v8h*)(arow1 + kk + 16 + hb * 8);
    bm.h[0] = *(const v8h*)(brow + kk + hb * 8);
    bm.h[1] = *(const v8h*)(brow + kk + 16 + hb * 8);
    acc0 = wmma_f16(a0.v, bm.v, acc0);
    acc1 = wmma_f16(a1.v, bm.v, acc1);
  }

#pragma unroll
  for (int r = 0; r < 8; ++r) {
    int o0 = obase + r + 8 * hb;
    int o1 = o0 + 16;
    out[((size_t)b * CCH + o0) * NTOK + nbase + l15] = acc0[r] + bo[o0];
    out[((size_t)b * CCH + o1) * NTOK + nbase + l15] = acc1[r] + bo[o1];
  }
}

// --------------------------- launcher ---------------------------------------
extern "C" void kernel_launch(void* const* d_in, const int* in_sizes, int n_in,
                              void* d_out, int out_size, void* d_ws, size_t ws_size,
                              hipStream_t stream) {
  const float* x  = (const float*)d_in[0];
  const float* Wq = (const float*)d_in[1];
  const float* bq = (const float*)d_in[2];
  const float* Wk = (const float*)d_in[3];
  const float* bk = (const float*)d_in[4];
  const float* Wv = (const float*)d_in[5];
  const float* bv = (const float*)d_in[6];
  const float* Wo = (const float*)d_in[7];
  const float* bo = (const float*)d_in[8];
  float* out = (float*)d_out;

  // workspace carve-up (all f16), ~12.3 MB total
  const size_t WSZ  = (size_t)CCH * CCH;        // 65536
  const size_t MSZ  = (size_t)BATCH * NTOK * CCH;
  _Float16* W16 = (_Float16*)d_ws;              // Wq|Wk|Wv|Wo  (4*WSZ)
  _Float16* Xt  = W16 + 4 * WSZ;                // [b][n][c]
  _Float16* Qb  = Xt + MSZ;                     // [b][n][c] (pre-scaled)
  _Float16* Kb  = Qb + MSZ;                     // [b][n][c]
  _Float16* Vb  = Kb + MSZ;                     // [b][c][n]
  _Float16* Ob  = Vb + MSZ;                     // [b][n][c]

  csa_cvt_weights<<<(4 * WSZ) / 256, 256, 0, stream>>>(Wq, Wk, Wv, Wo, W16);
  csa_transpose_x<<<MSZ / 256, 256, 0, stream>>>(x, Xt);

  // QKV: BATCH * 3 * (C/32) * (N/16) = 6912 waves, 8 waves per 256-thr block
  csa_qkv_proj<<<6912 / 8, 256, 0, stream>>>(Xt, W16, bq, bk, bv, Qb, Kb, Vb);

  // attention: grid (N/64, B*heads), 4 waves per block
  csa_attention<<<dim3(NTOK / 64, BATCH * NHEAD), 128, 0, stream>>>(Qb, Kb, Vb, Ob);

  // output projection: BATCH * (C/32) * (N/16) = 2304 waves
  csa_out_proj<<<2304 / 8, 256, 0, stream>>>(Ob, W16 + 3 * WSZ, bo, out);
}
